// OptimizedMOE_11390253269275
// MI455X (gfx1250) — matrix operand order repaired and checked by reading, not compiled, not run on hardware
//
#include <hip/hip_runtime.h>
#include <math.h>

typedef __attribute__((ext_vector_type(16))) __bf16 v16bf;
typedef __attribute__((ext_vector_type(8)))  float  v8f;

#define C_IN   256
#define HIDC   512
#define OUT_C  256
#define NEXP   4
#define HW     6400
#define TILE_N 64
#define XSTR   264   // bf16 elems/row of xT (fallback path): 16B aligned fragments
#define HSTR   520   // bf16 elems/row of hT: 1040B rows -> 16B aligned fragments

__device__ __forceinline__ float silu_f(float x) {
    return x / (1.0f + __expf(-x));
}

// ---- packed-weight A-fragment load: [mtile][kstep][lane][16 bf16], 2x b128 ----
__device__ __forceinline__ v16bf load_a_p(const unsigned* __restrict__ P, int ksteps,
                                          int mtile, int kstep, int lane) {
    const uint4* q = (const uint4*)(P + (((size_t)mtile * ksteps + kstep) * 32 + lane) * 8);
    union { v16bf v; uint4 q[2]; } t;
    t.q[0] = q[0];
    t.q[1] = q[1];
    return t.v;
}

// ---- B-fragment from [row=n][k] bf16 buffer (row stride ldr elems): 2x b128 ----
// lanes 0-15 carry K=0..15, lanes 16-31 carry K=16..31 (contiguous per lane).
__device__ __forceinline__ v16bf load_b_gt(const __bf16* __restrict__ base, int ldr,
                                           int k0, int lane) {
    const uint4* q = (const uint4*)(base + (size_t)(lane & 15) * ldr
                                        + k0 + (lane >> 4) * 16);
    union { v16bf v; uint4 q[2]; } t;
    t.q[0] = q[0];
    t.q[1] = q[1];
    return t.v;
}

// ---------------- weight fragment pre-pack (fp32 row-major -> bf16 fragments) --------
extern "C" __global__ __launch_bounds__(256)
void pack_frag(const float* __restrict__ W, unsigned* __restrict__ P,
               int ksteps, int K, int ndw) {
    int tid = blockIdx.x * 256 + threadIdx.x;
    if (tid >= ndw) return;
    int v     = tid & 7;
    int lane  = (tid >> 3) & 31;
    int rest  = tid >> 8;
    int kstep = rest % ksteps;
    int mtile = rest / ksteps;
    int m  = mtile * 16 + (lane & 15);
    int kh = (lane >> 4) * 8;
    int ko = 2 * kh + 2 * v;
    const float* s = W + (size_t)m * K + kstep * 32 + ko;
    union { __bf16 h[2]; unsigned u; } p;
    p.h[0] = (__bf16)s[0];
    p.h[1] = (__bf16)s[1];
    P[tid] = p.u;
}

// ---------------- x transpose+convert pass: [B,C,HW] fp32 -> [B,HW,C] bf16 -----------
extern "C" __global__ __launch_bounds__(256)
void xpose_bf16(const float* __restrict__ x, __bf16* __restrict__ xt) {
    __shared__ __bf16 tile[64][65];
    int pc = blockIdx.x;                 // pixel block (0..99)
    int cc = blockIdx.y;                 // channel block (0..3)
    int bb = blockIdx.z;                 // batch
    int tid = threadIdx.x;
    int lo4 = tid >> 6;                  // 0..3
    {   // read 64 channels x 64 pixels, pixel-contiguous (coalesced)
        int p = tid & 63;
#pragma unroll
        for (int it = 0; it < 16; ++it) {
            int c = it * 4 + lo4;
            tile[p][c] = (__bf16)x[((size_t)(bb * 256 + cc * 64 + c)) * HW + pc * 64 + p];
        }
    }
    __syncthreads();
    {   // write 64 pixels x 64 channels, channel-contiguous (coalesced)
        int c = tid & 63;
#pragma unroll
        for (int it = 0; it < 16; ++it) {
            int p = it * 4 + lo4;
            xt[((size_t)(bb * HW + pc * 64 + p)) * C_IN + cc * 64 + c] = tile[p][c];
        }
    }
}

// ---------------- router stage 0: 4x4 avg pool (+ zero glog accumulator) --------------
extern "C" __global__ __launch_bounds__(256)
void moe_pool(const float* __restrict__ x, float* __restrict__ xd, float* __restrict__ glog) {
    int idx = blockIdx.x * 256 + threadIdx.x;
    if (idx < 32) glog[idx] = 0.0f;
    if (idx >= 8 * 256 * 20 * 20) return;
    int j = idx % 20;
    int i = (idx / 20) % 20;
    int c = (idx / 400) % 256;
    int b = idx / 102400;
    const float* p = x + ((size_t)(b * 256 + c) * 80 + i * 4) * 80 + j * 4;
    float s = 0.0f;
#pragma unroll
    for (int di = 0; di < 4; ++di)
#pragma unroll
        for (int dj = 0; dj < 4; ++dj) s += p[di * 80 + dj];
    xd[idx] = s * (1.0f / 16.0f);
}

// ---------------- router stage 1: conv3x3 + affine + silu + 1x1, accumulate mean logits
extern "C" __global__ __launch_bounds__(256)
void moe_router(const float* __restrict__ xd, const float* __restrict__ rw1,
                const float* __restrict__ rs1, const float* __restrict__ rb1,
                const float* __restrict__ rw2, float* __restrict__ glog) {
    __shared__ float red[256];
    __shared__ float hbuf[32];
    int tid = threadIdx.x;
    int pos = blockIdx.x % 400;
    int b   = blockIdx.x / 400;
    int i = pos / 20, j = pos % 20;
    int oc = tid & 31, g = tid >> 5;
    float partial = 0.0f;
    for (int cc = 0; cc < 32; ++cc) {
        int c = g * 32 + cc;
        const float* xp = xd + (size_t)(b * 256 + c) * 400;
        const float* wp = rw1 + ((size_t)oc * 256 + c) * 9;
#pragma unroll
        for (int kh = 0; kh < 3; ++kh) {
            int ii = i + kh - 1;
            if (ii < 0 || ii >= 20) continue;
#pragma unroll
            for (int kw = 0; kw < 3; ++kw) {
                int jj = j + kw - 1;
                if (jj < 0 || jj >= 20) continue;
                partial += xp[ii * 20 + jj] * wp[kh * 3 + kw];
            }
        }
    }
    red[oc * 8 + g] = partial;
    __syncthreads();
    if (tid < 32) {
        float s = 0.0f;
#pragma unroll
        for (int k = 0; k < 8; ++k) s += red[tid * 8 + k];
        hbuf[tid] = silu_f(s * rs1[tid] + rb1[tid]);
    }
    __syncthreads();
    if (tid < 4) {
        float d = 0.0f;
#pragma unroll
        for (int k = 0; k < 32; ++k) d += hbuf[k] * rw2[tid * 32 + k];
        atomicAdd(&glog[b * 4 + tid], d * (1.0f / 400.0f));
    }
}

// ---------------- router stage 2: softmax, top-2, renormalize -> cw[b,e] --------------
extern "C" __global__ void moe_gate(const float* __restrict__ glog,
                                    const float* __restrict__ rs2,
                                    const float* __restrict__ rb2,
                                    float* __restrict__ cw) {
    int b = threadIdx.x;
    if (b >= 8) return;
    float g[4], p[4];
    float m = -1e30f;
#pragma unroll
    for (int e = 0; e < 4; ++e) { g[e] = rs2[e] * glog[b * 4 + e] + rb2[e]; m = fmaxf(m, g[e]); }
    float s = 0.0f;
#pragma unroll
    for (int e = 0; e < 4; ++e) { p[e] = __expf(g[e] - m); s += p[e]; }
#pragma unroll
    for (int e = 0; e < 4; ++e) p[e] /= s;
    int i1 = 0;
#pragma unroll
    for (int e = 1; e < 4; ++e) if (p[e] > p[i1]) i1 = e;   // strict > : first-occurrence ties
    int i2 = (i1 == 0) ? 1 : 0;
#pragma unroll
    for (int e = 0; e < 4; ++e) if (e != i1 && p[e] > p[i2]) i2 = e;
    float denom = p[i1] + p[i2] + 1e-6f;
#pragma unroll
    for (int e = 0; e < 4; ++e) cw[b * 4 + e] = 0.0f;
    cw[b * 4 + i1] = p[i1] / denom;
    cw[b * 4 + i2] = p[i2] / denom;
}

// ======================================================================================
// Main fused kernel, preferred path: x pre-transposed to [B,HW,C] bf16 in global.
// B fragments come straight from global (2x b128); LDS holds only the hid tile.
// ======================================================================================
extern "C" __global__ __launch_bounds__(256)
void moe_main_xt(const __bf16* __restrict__ xt,
                 const unsigned* __restrict__ psw,
                 const unsigned* __restrict__ pew1,
                 const unsigned* __restrict__ pew2,
                 const float* __restrict__ es1, const float* __restrict__ eb1,
                 const float* __restrict__ es2, const float* __restrict__ eb2,
                 const float* __restrict__ ss,  const float* __restrict__ sb,
                 const float* __restrict__ cw,  float* __restrict__ out) {
    extern __shared__ char smem_raw[];
    __bf16* hT = (__bf16*)smem_raw;            // [TILE_N][HSTR]

    const int tid  = threadIdx.x;
    const int lane = tid & 31;
    const int w    = tid >> 5;                 // 8 wave32s
    const int b    = blockIdx.y;
    const int col0 = blockIdx.x * TILE_N;

    const int nt   = w & 3;                    // column tile 0..3
    const int mh   = w >> 2;                   // row-tile group 0/1
    const int ncol = nt * 16;

    // ---- x B-fragments straight from global, cached in registers (64 VGPRs);
    //      reused by the shared expert and every expert's GEMM1 ----
    const __bf16* xbase = xt + ((size_t)b * HW + col0 + ncol) * C_IN;
    v16bf xB[8];
#pragma unroll
    for (int k = 0; k < 8; ++k) xB[k] = load_b_gt(xbase, C_IN, k * 32, lane);

    // ---- shared expert: out_sh = silu(affine(sw @ x)) ----
    v8f acc[8];
#pragma unroll
    for (int i = 0; i < 8; ++i) acc[i] = (v8f){};
#pragma unroll
    for (int i = 0; i < 8; ++i) {
        int mt = mh * 8 + i;
#pragma unroll
        for (int k = 0; k < 8; ++k) {
            v16bf am = load_a_p(psw, 8, mt, k, lane);
            acc[i] = __builtin_amdgcn_wmma_f32_16x16x32_bf16(
                false, am, false, xB[k], (short)0, acc[i], false, false);
        }
    }
#pragma unroll
    for (int i = 0; i < 8; ++i) {
        int rb = (mh * 8 + i) * 16 + (lane >> 4) * 8;
#pragma unroll
        for (int r = 0; r < 8; ++r) {
            float vv = acc[i][r] * ss[rb + r] + sb[rb + r];
            acc[i][r] = silu_f(vv);
        }
    }

    // ---- experts (block-uniform skip of cw==0) ----
    const __bf16* hbase = hT + (size_t)ncol * HSTR;
    for (int e = 0; e < NEXP; ++e) {
        float cwv = cw[b * NEXP + e];
        __syncthreads();                       // previous hT reads complete
        if (cwv != 0.0f) {
            // GEMM1: hid = silu(affine(ew1[e] @ xtile)) -> hT (bf16, [pix][hid])
            for (int i = 0; i < 16; ++i) {
                int mt = mh * 16 + i;
                v8f t = (v8f){};
#pragma unroll
                for (int k = 0; k < 8; ++k) {
                    v16bf am = load_a_p(pew1, 8, e * 32 + mt, k, lane);
                    t = __builtin_amdgcn_wmma_f32_16x16x32_bf16(
                        false, am, false, xB[k], (short)0, t, false, false);
                }
                int hb = mt * 16 + (lane >> 4) * 8;   // mult of 8 -> 16B aligned
                union { uint4 q; __bf16 h[8]; } pk;
#pragma unroll
                for (int r = 0; r < 8; ++r) {
                    float vv = t[r] * es1[e * HIDC + hb + r] + eb1[e * HIDC + hb + r];
                    pk.h[r] = (__bf16)silu_f(vv);
                }
                int pcol = ncol + (lane & 15);
                *(uint4*)(hT + (size_t)pcol * HSTR + hb) = pk.q;
            }
            __syncthreads();
            // GEMM2: acc += cw * affine(ew2[e] @ hid)
#pragma unroll
            for (int i = 0; i < 8; ++i) {
                int mt = mh * 8 + i;
                v8f t = (v8f){};
#pragma unroll
                for (int k = 0; k < 16; ++k) {
                    v16bf am = load_a_p(pew2, 16, e * 16 + mt, k, lane);
                    v16bf bm = load_b_gt(hbase, HSTR, k * 32, lane);
                    t = __builtin_amdgcn_wmma_f32_16x16x32_bf16(
                        false, am, false, bm, (short)0, t, false, false);
                }
                int rb = mt * 16 + (lane >> 4) * 8;
#pragma unroll
                for (int r = 0; r < 8; ++r) {
                    acc[i][r] += cwv * (t[r] * es2[e * OUT_C + rb + r]
                                        + eb2[e * OUT_C + rb + r]);
                }
            }
        }
    }

    // ---- store ----
    int pcol = col0 + ncol + (lane & 15);
    float* ob = out + (size_t)b * OUT_C * HW;
#pragma unroll
    for (int i = 0; i < 8; ++i) {
        int rb = (mh * 8 + i) * 16 + (lane >> 4) * 8;
#pragma unroll
        for (int r = 0; r < 8; ++r) {
            ob[(size_t)(rb + r) * HW + pcol] = acc[i][r];
        }
    }
}

// ======================================================================================
// Fallback path (small workspace): stage fp32 x tile through LDS per block (round-2).
// ======================================================================================
extern "C" __global__ __launch_bounds__(256)
void moe_main_lds(const float* __restrict__ x,
                  const unsigned* __restrict__ psw,
                  const unsigned* __restrict__ pew1,
                  const unsigned* __restrict__ pew2,
                  const float* __restrict__ es1, const float* __restrict__ eb1,
                  const float* __restrict__ es2, const float* __restrict__ eb2,
                  const float* __restrict__ ss,  const float* __restrict__ sb,
                  const float* __restrict__ cw,  float* __restrict__ out) {
    extern __shared__ char smem_raw[];
    __bf16* hT = (__bf16*)smem_raw;            // [TILE_N][HSTR]
    __bf16* xT = (__bf16*)smem_raw;            // [TILE_N][XSTR] alias, pre-expert only

    const int tid  = threadIdx.x;
    const int lane = tid & 31;
    const int w    = tid >> 5;
    const int b    = blockIdx.y;
    const int col0 = blockIdx.x * TILE_N;

    const float* xb = x + (size_t)b * C_IN * HW + col0;
#pragma unroll 4
    for (int it = 0; it < (C_IN * TILE_N) / 256; ++it) {
        int idx = it * 256 + tid;
        int c = idx >> 6;
        int p = idx & 63;
        xT[p * XSTR + c] = (__bf16)xb[(size_t)c * HW + p];
    }
    __syncthreads();

    const int nt   = w & 3;
    const int mh   = w >> 2;
    const int ncol = nt * 16;

    v16bf xB[8];
#pragma unroll
    for (int k = 0; k < 8; ++k) xB[k] = load_b_gt(xT + (size_t)ncol * XSTR, XSTR, k * 32, lane);

    v8f acc[8];
#pragma unroll
    for (int i = 0; i < 8; ++i) acc[i] = (v8f){};
#pragma unroll
    for (int i = 0; i < 8; ++i) {
        int mt = mh * 8 + i;
#pragma unroll
        for (int k = 0; k < 8; ++k) {
            v16bf am = load_a_p(psw, 8, mt, k, lane);
            acc[i] = __builtin_amdgcn_wmma_f32_16x16x32_bf16(
                false, am, false, xB[k], (short)0, acc[i], false, false);
        }
    }
#pragma unroll
    for (int i = 0; i < 8; ++i) {
        int rb = (mh * 8 + i) * 16 + (lane >> 4) * 8;
#pragma unroll
        for (int r = 0; r < 8; ++r) {
            float vv = acc[i][r] * ss[rb + r] + sb[rb + r];
            acc[i][r] = silu_f(vv);
        }
    }

    const __bf16* hbase = hT + (size_t)ncol * HSTR;
    for (int e = 0; e < NEXP; ++e) {
        float cwv = cw[b * NEXP + e];
        __syncthreads();
        if (cwv != 0.0f) {
            for (int i = 0; i < 16; ++i) {
                int mt = mh * 16 + i;
                v8f t = (v8f){};
#pragma unroll
                for (int k = 0; k < 8; ++k) {
                    v16bf am = load_a_p(pew1, 8, e * 32 + mt, k, lane);
                    t = __builtin_amdgcn_wmma_f32_16x16x32_bf16(
                        false, am, false, xB[k], (short)0, t, false, false);
                }
                int hb = mt * 16 + (lane >> 4) * 8;
                union { uint4 q; __bf16 h[8]; } pk;
#pragma unroll
                for (int r = 0; r < 8; ++r) {
                    float vv = t[r] * es1[e * HIDC + hb + r] + eb1[e * HIDC + hb + r];
                    pk.h[r] = (__bf16)silu_f(vv);
                }
                int pcol = ncol + (lane & 15);
                *(uint4*)(hT + (size_t)pcol * HSTR + hb) = pk.q;
            }
            __syncthreads();
#pragma unroll
            for (int i = 0; i < 8; ++i) {
                int mt = mh * 8 + i;
                v8f t = (v8f){};
#pragma unroll
                for (int k = 0; k < 16; ++k) {
                    v16bf am = load_a_p(pew2, 16, e * 16 + mt, k, lane);
                    v16bf bm = load_b_gt(hbase, HSTR, k * 32, lane);
                    t = __builtin_amdgcn_wmma_f32_16x16x32_bf16(
                        false, am, false, bm, (short)0, t, false, false);
                }
                int rb = mt * 16 + (lane >> 4) * 8;
#pragma unroll
                for (int r = 0; r < 8; ++r) {
                    acc[i][r] += cwv * (t[r] * es2[e * OUT_C + rb + r]
                                        + eb2[e * OUT_C + rb + r]);
                }
            }
        }
    }

    int pcol = col0 + ncol + (lane & 15);
    float* ob = out + (size_t)b * OUT_C * HW;
#pragma unroll
    for (int i = 0; i < 8; ++i) {
        int rb = (mh * 8 + i) * 16 + (lane >> 4) * 8;
#pragma unroll
        for (int r = 0; r < 8; ++r) {
            ob[(size_t)(rb + r) * HW + pcol] = acc[i][r];
        }
    }
}

extern "C" void kernel_launch(void* const* d_in, const int* in_sizes, int n_in,
                              void* d_out, int out_size, void* d_ws, size_t ws_size,
                              hipStream_t stream) {
    const float* x   = (const float*)d_in[0];
    const float* rw1 = (const float*)d_in[1];
    const float* rs1 = (const float*)d_in[2];
    const float* rb1 = (const float*)d_in[3];
    const float* rw2 = (const float*)d_in[4];
    const float* rs2 = (const float*)d_in[5];
    const float* rb2 = (const float*)d_in[6];
    const float* ew1 = (const float*)d_in[7];
    const float* es1 = (const float*)d_in[8];
    const float* eb1 = (const float*)d_in[9];
    const float* ew2 = (const float*)d_in[10];
    const float* es2 = (const float*)d_in[11];
    const float* eb2 = (const float*)d_in[12];
    const float* sw  = (const float*)d_in[13];
    const float* ss  = (const float*)d_in[14];
    const float* sb  = (const float*)d_in[15];
    float* out = (float*)d_out;

    // ---- workspace layout ----
    float*    xd   = (float*)d_ws;                 // 819200 floats
    float*    glog = xd + 819200;                  // 32
    float*    cwb  = glog + 32;                    // 32
    unsigned* psw  = (unsigned*)(cwb + 32);        // 32768 dwords
    unsigned* pew1 = psw + 32768;                  // 262144 dwords
    unsigned* pew2 = pew1 + 262144;                // 262144 dwords
    __bf16*   xt   = (__bf16*)(pew2 + 262144);     // 8*6400*256 bf16 = 26.2 MB
    const size_t need_xt = (size_t)819264 * 4 + (size_t)557056 * 4
                         + (size_t)8 * HW * C_IN * 2;   // ~31.7 MB

    // ---- weight fragment packing (deterministic, every launch) ----
    pack_frag<<<(32768  + 255) / 256, 256, 0, stream>>>(sw,  psw,  8,  256, 32768);
    pack_frag<<<(262144 + 255) / 256, 256, 0, stream>>>(ew1, pew1, 8,  256, 262144);
    pack_frag<<<(262144 + 255) / 256, 256, 0, stream>>>(ew2, pew2, 16, 512, 262144);

    // ---- router ----
    moe_pool<<<3200, 256, 0, stream>>>(x, xd, glog);
    moe_router<<<3200, 256, 0, stream>>>(xd, rw1, rs1, rb1, rw2, glog);
    moe_gate<<<1, 32, 0, stream>>>(glog, rs2, rb2, cwb);

    // ---- main fused GEMM ----
    size_t smem = (size_t)TILE_N * HSTR * 2;       // 66560 B LDS
    dim3 grid(HW / TILE_N, 8);
    if (ws_size >= need_xt) {
        dim3 tg(HW / 64, C_IN / 64, 8);
        xpose_bf16<<<tg, 256, 0, stream>>>(x, xt);
        moe_main_xt<<<grid, 256, smem, stream>>>(xt, psw, pew1, pew2,
                                                 es1, eb1, es2, eb2, ss, sb, cwb, out);
    } else {
        moe_main_lds<<<grid, 256, smem, stream>>>(x, psw, pew1, pew2,
                                                  es1, eb1, es2, eb2, ss, sb, cwb, out);
    }
}
